// RecurrentAgent_72310069395977
// MI455X (gfx1250) — 16-bit and 8-bit matrix operand order repaired and checked by reading, compiled, not run on hardware
//
#include <hip/hip_runtime.h>
#include <hip/hip_bf16.h>


typedef __attribute__((ext_vector_type(16))) _Float16 v16h;
typedef __attribute__((ext_vector_type(8)))  float    v8f;
typedef __attribute__((ext_vector_type(8)))  int      v8i;

#define TT 128
#define BB 16
#define HH 256
#define NIMG 2048


__device__ __forceinline__ v16h load_a_frag(const _Float16* A, int lda) {
  int lane = threadIdx.x & 31;
  int m = lane & 15;
  int kb = (lane >> 4) << 3;
  const _Float16* p = A + m * lda;
  v16h a;
#pragma unroll
  for (int j = 0; j < 8; ++j) { a[j] = p[kb + j]; a[8 + j] = p[16 + kb + j]; }
  return a;
}

__device__ __forceinline__ v16h load_b_frag(const _Float16* B, int ldb) {
  int lane = threadIdx.x & 31;
  int n = lane & 15;
  int kb = (lane >> 4) << 3;
  v16h b;
#pragma unroll
  for (int j = 0; j < 8; ++j) { b[j] = B[(kb + j) * ldb + n]; b[8 + j] = B[(16 + kb + j) * ldb + n]; }
  return b;
}


__device__ __forceinline__ unsigned char f32_to_e4m3(float x) {
  unsigned u = __float_as_uint(x);
  unsigned char s = (unsigned char)((u >> 24) & 0x80u);
  unsigned a = u & 0x7FFFFFFFu;
  if (a >= 0x7F800000u) return s;
  int e8 = (int)(a >> 23) - 127 + 7;
  if (e8 <= 0) {
    int q = __float2int_rn(__uint_as_float(a) * 512.0f);
    if (q >= 8) return (unsigned char)(s | 0x08);
    return (unsigned char)(s | (unsigned)q);
  }
  unsigned mant = a & 0x7FFFFFu;
  unsigned mr = mant + 0x7FFFFu + ((mant >> 20) & 1u);
  unsigned m3 = mr >> 20;
  if (m3 >= 8u) { m3 = 0u; ++e8; }
  if (e8 > 15 || (e8 == 15 && m3 == 7u)) return (unsigned char)(s | 0x7E);
  return (unsigned char)(s | ((unsigned)e8 << 3) | m3);
}

__device__ __forceinline__ float sigm(float x) { return 1.0f / (1.0f + __expf(-x)); }

__global__ __launch_bounds__(256) void conv1_kernel(
    const float* __restrict__ img, const float* __restrict__ w1,
    const float* __restrict__ b1, _Float16* __restrict__ out) {
  extern __shared__ char smem[];
  _Float16* Aim = (_Float16*)smem;
  _Float16* Bw  = (_Float16*)(smem + 400 * 256 * 2);
  int n = blockIdx.x, tid = threadIdx.x;
  const float inv255 = 1.0f / 255.0f;
  for (int i = tid; i < 400 * 256; i += 256) {
    int k = i / 400, p = i - k * 400;
    int ic = k & 3, sp = k >> 2, ky = sp >> 3, kx = sp & 7;
    int oy = p / 20, ox = p - oy * 20;
    int iy = oy * 4 + ky, ix = ox * 4 + kx;
    Aim[p * 256 + k] =
        (_Float16)(img[(((long)n * 4 + ic) * 84 + iy) * 84 + ix] * inv255);
  }
  for (int i = tid; i < 256 * 32; i += 256) {
    int k = i >> 5, o = i & 31;
    int ic = k & 3, sp = k >> 2, ky = sp >> 3, kx = sp & 7;
    Bw[i] = (_Float16)w1[((o * 4 + ic) * 8 + ky) * 8 + kx];
  }
  __syncthreads();
  int wave = tid >> 5, lane = tid & 31, nn = lane & 15, g = lane >> 4;
  for (int tile = wave; tile < 50; tile += 8) {
    int mt = tile >> 1, nt = tile & 1;
    v8f acc = {};
#pragma unroll
    for (int kc = 0; kc < 8; ++kc) {
      v16h a = load_a_frag(&Aim[(mt * 16) * 256 + kc * 32], 256);
      v16h b = load_b_frag(&Bw[(kc * 32) * 32 + nt * 16], 32);
      acc = __builtin_amdgcn_wmma_f32_16x16x32_f16(false, a, false, b,
                                                   (short)0, acc, false, false);
    }
    int o = nt * 16 + nn;
    float bias = b1[o];
#pragma unroll
    for (int r = 0; r < 8; ++r) {
      int p = mt * 16 + r + 8 * g;
      float v = fmaxf(acc[r] + bias, 0.0f);
      out[((long)n * 400 + p) * 32 + o] = (_Float16)v;
    }
  }
}

__global__ __launch_bounds__(256) void conv2_kernel(
    const _Float16* __restrict__ in, const float* __restrict__ w2,
    const float* __restrict__ b2, _Float16* __restrict__ out) {
  extern __shared__ char smem[];
  _Float16* Aim = (_Float16*)smem;
  _Float16* Bw  = (_Float16*)(smem + 96 * 512 * 2);
  int n = blockIdx.x, tid = threadIdx.x;
  for (int i = tid; i < 96 * 512; i += 256) {
    int p = i >> 9, k = i & 511;
    _Float16 v = (_Float16)0.0f;
    if (p < 81) {
      int c = k & 31, sp = k >> 5, ky = sp >> 2, kx = sp & 3;
      int oy = p / 9, ox = p - oy * 9;
      int iy = oy * 2 + ky, ix = ox * 2 + kx;
      v = in[((long)n * 400 + iy * 20 + ix) * 32 + c];
    }
    Aim[i] = v;
  }
  for (int i = tid; i < 512 * 64; i += 256) {
    int k = i >> 6, o = i & 63;
    int ic = k & 31, sp = k >> 5, ky = sp >> 2, kx = sp & 3;
    Bw[i] = (_Float16)w2[((o * 32 + ic) * 4 + ky) * 4 + kx];
  }
  __syncthreads();
  int wave = tid >> 5, lane = tid & 31, nn = lane & 15, g = lane >> 4;
  for (int tile = wave; tile < 24; tile += 8) {
    int mt = tile >> 2, nt = tile & 3;
    v8f acc = {};
#pragma unroll
    for (int kc = 0; kc < 16; ++kc) {
      v16h a = load_a_frag(&Aim[(mt * 16) * 512 + kc * 32], 512);
      v16h b = load_b_frag(&Bw[(kc * 32) * 64 + nt * 16], 64);
      acc = __builtin_amdgcn_wmma_f32_16x16x32_f16(false, a, false, b,
                                                   (short)0, acc, false, false);
    }
    int o = nt * 16 + nn;
    float bias = b2[o];
#pragma unroll
    for (int r = 0; r < 8; ++r) {
      int p = mt * 16 + r + 8 * g;
      if (p < 81) {
        float v = fmaxf(acc[r] + bias, 0.0f);
        out[((long)n * 81 + p) * 64 + o] = (_Float16)v;
      }
    }
  }
}

__global__ __launch_bounds__(256) void conv3_kernel(
    const _Float16* __restrict__ in, const float* __restrict__ w3,
    const float* __restrict__ b3, _Float16* __restrict__ out) {
  extern __shared__ char smem[];
  _Float16* Aim = (_Float16*)smem;
  _Float16* Bw  = (_Float16*)(smem + 64 * 576 * 2);
  int n = blockIdx.x, tid = threadIdx.x;
  for (int i = tid; i < 64 * 576; i += 256) {
    int p = i / 576, k = i - p * 576;
    _Float16 v = (_Float16)0.0f;
    if (p < 49) {
      int c = k & 63, sp = k >> 6, ky = sp / 3, kx = sp - ky * 3;
      int oy = p / 7, ox = p - oy * 7;
      int iy = oy + ky, ix = ox + kx;
      v = in[((long)n * 81 + iy * 9 + ix) * 64 + c];
    }
    Aim[i] = v;
  }
  for (int i = tid; i < 576 * 64; i += 256) {
    int k = i >> 6, o = i & 63;
    int c = k & 63, sp = k >> 6, ky = sp / 3, kx = sp - ky * 3;
    Bw[i] = (_Float16)w3[((o * 64 + c) * 3 + ky) * 3 + kx];
  }
  __syncthreads();
  int wave = tid >> 5, lane = tid & 31, nn = lane & 15, g = lane >> 4;
  for (int tile = wave; tile < 16; tile += 8) {
    int mt = tile >> 2, nt = tile & 3;
    v8f acc = {};
#pragma unroll
    for (int kc = 0; kc < 18; ++kc) {
      v16h a = load_a_frag(&Aim[(mt * 16) * 576 + kc * 32], 576);
      v16h b = load_b_frag(&Bw[(kc * 32) * 64 + nt * 16], 64);
      acc = __builtin_amdgcn_wmma_f32_16x16x32_f16(false, a, false, b,
                                                   (short)0, acc, false, false);
    }
    int o = nt * 16 + nn;
    float bias = b3[o];
#pragma unroll
    for (int r = 0; r < 8; ++r) {
      int p = mt * 16 + r + 8 * g;
      if (p < 49)
        out[(long)n * 3136 + o * 49 + p] = (_Float16)fmaxf(acc[r] + bias, 0.0f);
    }
  }
}

template <bool RELU, bool OUT_HALF>
__global__ __launch_bounds__(256) void gemm_bt_kernel(
    const _Float16* __restrict__ A, int lda, const float* __restrict__ Bw,
    int ldb, const float* __restrict__ bias1, const float* __restrict__ bias2,
    void* __restrict__ outp, int ldo, int K) {
  extern __shared__ char smem[];
  _Float16* As = (_Float16*)smem;
  _Float16* Bs = (_Float16*)(smem + 64 * 32 * 2);
  int m0 = blockIdx.x * 64, n0 = blockIdx.y * 64;
  int tid = threadIdx.x, wave = tid >> 5;
  int tr = wave >> 1, tc0 = (wave & 1) * 2;
  v8f acc0 = {}, acc1 = {};
  for (int k0 = 0; k0 < K; k0 += 32) {
    for (int i = tid; i < 64 * 32; i += 256) {
      int r = i >> 5, kk = i & 31;
      As[i] = A[(long)(m0 + r) * lda + k0 + kk];
      if (k0 + 32 < K) __builtin_prefetch(&A[(long)(m0 + r) * lda + k0 + 32 + kk], 0, 1);
    }
    for (int i = tid; i < 64 * 32; i += 256) {
      int cc = i >> 5, kk = i & 31;
      Bs[kk * 64 + cc] = (_Float16)Bw[(long)(n0 + cc) * ldb + k0 + kk];
      if (k0 + 32 < K) __builtin_prefetch(&Bw[(long)(n0 + cc) * ldb + k0 + 32 + kk], 0, 1);
    }
    __syncthreads();
    v16h a  = load_a_frag(&As[tr * 16 * 32], 32);
    v16h b0 = load_b_frag(&Bs[tc0 * 16], 64);
    v16h b1 = load_b_frag(&Bs[(tc0 + 1) * 16], 64);
    acc0 = __builtin_amdgcn_wmma_f32_16x16x32_f16(false, a, false, b0, (short)0,
                                                  acc0, false, false);
    acc1 = __builtin_amdgcn_wmma_f32_16x16x32_f16(false, a, false, b1, (short)0,
                                                  acc1, false, false);
    __syncthreads();
  }
  int lane = tid & 31, nn = lane & 15, g = lane >> 4;
#pragma unroll
  for (int t2 = 0; t2 < 2; ++t2) {
    v8f acc = t2 ? acc1 : acc0;
    int nc = n0 + (tc0 + t2) * 16 + nn;
    float bsum = bias1[nc] + (bias2 ? bias2[nc] : 0.0f);
#pragma unroll
    for (int r = 0; r < 8; ++r) {
      long mr = m0 + tr * 16 + r + 8 * g;
      float v = acc[r] + bsum;
      if (RELU) v = fmaxf(v, 0.0f);
      if (OUT_HALF) ((_Float16*)outp)[mr * ldo + nc] = (_Float16)v;
      else          ((float*)outp)[mr * ldo + nc] = v;
    }
  }
}

__global__ __launch_bounds__(256) void prep_whh8_kernel(
    const float* __restrict__ w_hh, unsigned int* __restrict__ whh8) {
  int idx = blockIdx.x * 256 + threadIdx.x;
  if (idx >= 65536) return;
  int v = idx & 7, lane = (idx >> 3) & 31, ntG = (idx >> 8) & 63, kc = idx >> 14;
  int n = lane & 15, g = lane >> 4;
  int col = ntG * 16 + n;
  int j = col >> 2, gate = col & 3;
  long row = gate * 256 + j;
  unsigned int pack = 0;
#pragma unroll
  for (int b = 0; b < 4; ++b) {
    int K = kc * 64 + 32 * (v >> 2) + 16 * g + 4 * (v & 3) + b;
    pack |= ((unsigned int)f32_to_e4m3(w_hh[row * 256 + K])) << (8 * b);
  }
  whh8[idx] = pack;
}

__global__ __launch_bounds__(512) void lstm_kernel(
    const unsigned int* __restrict__ whh8, const float* __restrict__ xg,
    const int* __restrict__ done, const float* __restrict__ h0,
    const float* __restrict__ c0, float* __restrict__ hs,
    float* __restrict__ d_out) {
  extern __shared__ char smem[];
  unsigned int* wfrag = (unsigned int*)smem;
  unsigned char* h8 = (unsigned char*)(smem + 262144);
  int tid = threadIdx.x, wave = tid >> 5, lane = tid & 31;
  int nn = lane & 15, g = lane >> 4, gate = lane & 3, q = nn >> 2;

  for (int i = tid; i < 65536; i += 512) wfrag[i] = whh8[i];
  for (int i = tid; i < BB * HH; i += 512) h8[i] = f32_to_e4m3(h0[i]);

  v8f cst[4];
#pragma unroll
  for (int nt = 0; nt < 4; ++nt) {
    int j = wave * 16 + nt * 4 + q;
#pragma unroll
    for (int r = 0; r < 8; ++r) cst[nt][r] = c0[(r + 8 * g) * HH + j];
  }
  __syncthreads();

  for (int t = 0; t < TT; ++t) {
    for (int i = tid; i < BB * HH; i += 512)
      if (done[t * BB + (i >> 8)]) h8[i] = 0;
    __syncthreads();

    v8f acc[4];
#pragma unroll
    for (int nt = 0; nt < 4; ++nt) acc[nt] = 0.0f;

#pragma unroll
    for (int kc = 0; kc < 4; ++kc) {
      v8i a;
#pragma unroll
      for (int v = 0; v < 8; ++v) {
        int K = kc * 64 + 32 * (v >> 2) + 16 * g + 4 * (v & 3);
        a[v] = (int)*(const unsigned int*)&h8[nn * HH + K];
      }
#pragma unroll
      for (int nt = 0; nt < 4; ++nt) {
        const unsigned int* fb = &wfrag[(((long)kc * 64 + wave * 4 + nt) * 32 + lane) * 8];
        v8i b;
#pragma unroll
        for (int v = 0; v < 8; ++v) b[v] = (int)fb[v];
        acc[nt] = __builtin_amdgcn_wmma_f32_16x16x64_fp8_fp8(a, b, (short)0,
                                                             acc[nt], false, false);
      }
    }

#pragma unroll
    for (int nt = 0; nt < 4; ++nt) {
      int ccol = wave * 64 + nt * 16 + nn;
      int j = ccol >> 2;
      int orig_n = gate * 256 + j;
      int base = lane & ~3;
#pragma unroll
      for (int r = 0; r < 8; ++r) {
        int m = r + 8 * g;
        float gv = acc[nt][r] + xg[((long)t * BB + m) * 1024 + orig_n];
        float iv = __shfl(gv, base + 0, 32);
        float fv = __shfl(gv, base + 1, 32);
        float gg = __shfl(gv, base + 2, 32);
        float ov = __shfl(gv, base + 3, 32);
        float keep = done[t * BB + m] ? 0.0f : 1.0f;
        float cnew = sigm(fv) * (cst[nt][r] * keep) + sigm(iv) * tanhf(gg);
        float hnew = sigm(ov) * tanhf(cnew);
        cst[nt][r] = cnew;
        if (gate == 0) {
          hs[((long)t * BB + m) * HH + j] = hnew;
          h8[m * HH + j] = f32_to_e4m3(hnew);
        }
      }
    }
    __syncthreads();
  }

  if (gate == 0) {
#pragma unroll
    for (int nt = 0; nt < 4; ++nt) {
      int j = (wave * 64 + nt * 16 + nn) >> 2;
#pragma unroll
      for (int r = 0; r < 8; ++r)
        d_out[16384 + (r + 8 * g) * HH + j] = cst[nt][r];
    }
  }
}

__global__ __launch_bounds__(256) void heads_kernel(
    const float* __restrict__ hs, const float* __restrict__ pol_w,
    const float* __restrict__ pol_b, const float* __restrict__ val_w,
    const float* __restrict__ val_b, float* __restrict__ d_out) {
  int gid = blockIdx.x * 256 + threadIdx.x;
  int row = gid >> 3, slot = gid & 7;
  if (row < TT * BB) {
    const float* h = &hs[(long)row * HH];
    if (slot < 5) {
      float s = pol_b[slot];
      for (int k = 0; k < HH; ++k) s += h[k] * pol_w[slot * HH + k];
      d_out[row * 5 + slot] = s;
    } else if (slot == 5) {
      float s = val_b[0];
      for (int k = 0; k < HH; ++k) s += h[k] * val_w[k];
      d_out[TT * BB * 5 + row] = s;
    }
  }
  if (gid < BB * HH)
    d_out[TT * BB * 5 + TT * BB + gid] = hs[(long)(TT - 1) * BB * HH + gid];
}

extern "C" void kernel_launch(void* const* d_in, const int* in_sizes, int n_in,
                              void* d_out, int out_size, void* d_ws,
                              size_t ws_size, hipStream_t stream) {
  (void)in_sizes; (void)n_in; (void)out_size; (void)ws_size;
  const float* image = (const float*)d_in[0];
  const int*   done  = (const int*)d_in[1];
  const float* h0    = (const float*)d_in[2];
  const float* c0    = (const float*)d_in[3];
  const float* c1w = (const float*)d_in[4];  const float* c1b = (const float*)d_in[5];
  const float* c2w = (const float*)d_in[6];  const float* c2b = (const float*)d_in[7];
  const float* c3w = (const float*)d_in[8];  const float* c3b = (const float*)d_in[9];
  const float* fcw = (const float*)d_in[10]; const float* fcb = (const float*)d_in[11];
  const float* wih = (const float*)d_in[12]; const float* whh = (const float*)d_in[13];
  const float* bih = (const float*)d_in[14]; const float* bhh = (const float*)d_in[15];
  const float* polw = (const float*)d_in[16]; const float* polb = (const float*)d_in[17];
  const float* valw = (const float*)d_in[18]; const float* valb = (const float*)d_in[19];
  float* out = (float*)d_out;

  char* ws = (char*)d_ws;
  _Float16* ws1     = (_Float16*)(ws + 0);
  _Float16* ws2     = (_Float16*)(ws + 52428800);
  _Float16* feat3   = (_Float16*)(ws + 73662464);
  _Float16* feats16 = (_Float16*)(ws + 86507520);
  float*    xg      = (float*)(ws + 0);
  float*    hs      = (float*)(ws + 88604672);
  unsigned int* whh8 = (unsigned int*)(ws + 90701824);

  prep_whh8_kernel<<<256, 256, 0, stream>>>(whh, whh8);
  conv1_kernel<<<NIMG, 256, 400 * 256 * 2 + 256 * 32 * 2, stream>>>(image, c1w, c1b, ws1);
  conv2_kernel<<<NIMG, 256, 96 * 512 * 2 + 512 * 64 * 2, stream>>>(ws1, c2w, c2b, ws2);
  conv3_kernel<<<NIMG, 256, 64 * 576 * 2 + 576 * 64 * 2, stream>>>(ws2, c3w, c3b, feat3);
  gemm_bt_kernel<true, true><<<dim3(32, 8), 256, 8192, stream>>>(
      feat3, 3136, fcw, 3136, fcb, nullptr, feats16, 512, 3136);
  gemm_bt_kernel<false, false><<<dim3(32, 16), 256, 8192, stream>>>(
      feats16, 512, wih, 512, bih, bhh, xg, 1024, 512);
  lstm_kernel<<<1, 512, 262144 + 4096, stream>>>(whh8, xg, done, h0, c0, hs, out);
  heads_kernel<<<64, 256, 0, stream>>>(hs, polw, polb, valw, valb, out);
}
